// BachDuet_9122510536896
// MI455X (gfx1250) — matrix operand order repaired and checked by reading, not compiled, not run on hardware
//
#include <hip/hip_runtime.h>
#include <hip/hip_bf16.h>

// ---------------------------------------------------------------------------
// Types for CDNA5 WMMA (wave32): V_WMMA_F32_16X16X32_BF16
// ---------------------------------------------------------------------------
typedef __attribute__((ext_vector_type(16))) __bf16 v16bf;
typedef __attribute__((ext_vector_type(8)))  __bf16 v8bf;
typedef __attribute__((ext_vector_type(8)))  float  v8f;
typedef __attribute__((ext_vector_type(4)))  float  v4f;

#define TLEN  4096
#define HID   400
#define GATES 1600          // 4*HID
#define KPAD  416           // HID padded to multiple of 32 (13 K-tiles)
#define KT_H  13            // KPAD/32
#define NT_G  100           // GATES/16
#define EMB3  150           // 3*EMB
#define EMB3P 160           // padded to 5 K-tiles
#define KEY   300
#define STACKN 32

__device__ __forceinline__ float sigf(float x) { return 1.0f / (1.0f + __expf(-x)); }

// ---------------------------------------------------------------------------
// Pack a (G x Kdim) fp32 weight matrix into WMMA B-matrix lane layout, bf16.
// Tile (nt,kt) holds B[k][n] = W[n][k]. Lane layout for 32x16 bf16 B:
// lanes 0-15 hold K=0..7,16..23, lanes 16-31 hold K=8..15,24..31,
// N = lane%16. Zero padding out of range.
// ---------------------------------------------------------------------------
__global__ void pack_b_kernel(const float* __restrict__ W, int G, int Kdim,
                              int Ktiles, __bf16* __restrict__ out) {
  int nt = blockIdx.x, kt = blockIdx.y;
  int tid = threadIdx.x;              // 0..511
  int lane = tid >> 4, j = tid & 15;
  int k = kt * 32 + ((lane < 16) ? 0 : 8) + ((j < 8) ? j : (8 + j));
  int n = nt * 16 + (lane & 15);
  float v = (n < G && k < Kdim) ? W[(size_t)n * Kdim + k] : 0.0f;
  out[((size_t)(nt * Ktiles + kt)) * 512 + lane * 16 + j] = (__bf16)v;
}

// ---------------------------------------------------------------------------
// Embedding gather + concat -> bf16 activations, K padded 150->160 with zeros.
// ---------------------------------------------------------------------------
__global__ void embed_kernel(const int* __restrict__ xma, const int* __restrict__ xcpc,
                             const int* __restrict__ xr,
                             const float* __restrict__ em, const float* __restrict__ ec,
                             const float* __restrict__ er, __bf16* __restrict__ xbf) {
  int idx = blockIdx.x * blockDim.x + threadIdx.x;
  if (idx >= TLEN * EMB3P) return;
  int t = idx / EMB3P, e = idx % EMB3P;
  float v = 0.0f;
  if (e < 50)        v = em[xma[t] * 50 + e];
  else if (e < 100)  v = ec[xcpc[t] * 50 + (e - 50)];
  else if (e < 150)  v = er[xr[t] * 50 + (e - 100)];
  xbf[idx] = (__bf16)v;
}

// ---------------------------------------------------------------------------
// WMMA GEMM with 2x2 macro-tile per wave: C[32 x 32] per wave.
// 4 WMMAs per K-tile against 4 A-loads + 2 B-loads (B reused across M,
// A reused across N). grid=(Ntiles/2, Mtiles/2), block=32 (one wave).
// ---------------------------------------------------------------------------
__global__ void __launch_bounds__(32)
wmma_gemm_kernel(const __bf16* __restrict__ A, int lda,
                 const __bf16* __restrict__ Bp, const float* __restrict__ bias,
                 float* __restrict__ C, int ldc, int Ncols, int Ktiles) {
  const int lane = threadIdx.x;
  const int nt0 = blockIdx.x * 2, mt0 = blockIdx.y * 2;
  const int off = (lane >= 16) ? 8 : 0;   // A-matrix 16-bit lane K offset
  const __bf16* Arow0 = A + (size_t)(mt0 * 16 + (lane & 15)) * lda;
  const __bf16* Arow1 = Arow0 + (size_t)16 * lda;
  const __bf16* Bt0 = Bp + ((size_t)nt0 * Ktiles) * 512 + lane * 16;
  const __bf16* Bt1 = Bt0 + (size_t)Ktiles * 512;
  v8f acc00 = {}, acc01 = {}, acc10 = {}, acc11 = {};
  for (int kt = 0; kt < Ktiles; ++kt) {
    v8bf lo0 = *(const v8bf*)(Arow0 + kt * 32 + off);
    v8bf hi0 = *(const v8bf*)(Arow0 + kt * 32 + off + 16);
    v8bf lo1 = *(const v8bf*)(Arow1 + kt * 32 + off);
    v8bf hi1 = *(const v8bf*)(Arow1 + kt * 32 + off + 16);
    v16bf b0 = *(const v16bf*)(Bt0 + (size_t)kt * 512);
    v16bf b1 = *(const v16bf*)(Bt1 + (size_t)kt * 512);
    if (kt + 1 < Ktiles) {
      __builtin_prefetch(Bt0 + (size_t)(kt + 1) * 512, 0, 3);
      __builtin_prefetch(Bt1 + (size_t)(kt + 1) * 512, 0, 3);
    }
    v16bf a0, a1;
#pragma unroll
    for (int i = 0; i < 8; ++i) {
      a0[i] = lo0[i]; a0[8 + i] = hi0[i];
      a1[i] = lo1[i]; a1[8 + i] = hi1[i];
    }
    acc00 = __builtin_amdgcn_wmma_f32_16x16x32_bf16(false, a0, false, b0, (short)0, acc00, false, false);
    acc01 = __builtin_amdgcn_wmma_f32_16x16x32_bf16(false, a0, false, b1, (short)0, acc01, false, false);
    acc10 = __builtin_amdgcn_wmma_f32_16x16x32_bf16(false, a1, false, b0, (short)0, acc10, false, false);
    acc11 = __builtin_amdgcn_wmma_f32_16x16x32_bf16(false, a1, false, b1, (short)0, acc11, false, false);
  }
  const int rb = (lane >= 16) ? 8 : 0;   // D: VGPR v holds M=v (+8 for hi lanes)
  const int c0 = nt0 * 16 + (lane & 15), c1 = c0 + 16;
  const int r0 = mt0 * 16 + rb,          r1 = r0 + 16;
  if (c0 < Ncols) {
    float bv = bias[c0];
#pragma unroll
    for (int v = 0; v < 8; ++v) {
      C[(size_t)(r0 + v) * ldc + c0] = acc00[v] + bv;
      C[(size_t)(r1 + v) * ldc + c0] = acc10[v] + bv;
    }
  }
  if (c1 < Ncols) {
    float bv = bias[c1];
#pragma unroll
    for (int v = 0; v < 8; ++v) {
      C[(size_t)(r0 + v) * ldc + c1] = acc01[v] + bv;
      C[(size_t)(r1 + v) * ldc + c1] = acc11[v] + bv;
    }
  }
}

// ---------------------------------------------------------------------------
// Persistent single-workgroup LSTM scan. 1024 threads = 32 waves.
// 2-way K-split: wave (nw,half) computes K-half `half` of N-tiles
// nt = nw, nw+16, ... ; partial sums meet in LDS. This halves the dependent
// WMMA chain on the sequential critical path. h replicated across the 16
// A rows (D row 0 is the matvec result). Whh stays L2-resident.
// ---------------------------------------------------------------------------
__global__ void __launch_bounds__(1024)
lstm_scan_kernel(const float* __restrict__ xg,        // [T][1600] (includes bih)
                 const __bf16* __restrict__ WhhP,     // packed [100][13][512]
                 const float* __restrict__ bhh,       // [1600]
                 __bf16* __restrict__ outBF,          // [T][KPAD]
                 float* __restrict__ outF32) {        // [T][HID] or nullptr
  __shared__ __align__(16) __bf16 hbf[KPAD];
  __shared__ float gatesP[2][GATES];
  __shared__ float cst[HID];
  const int tid = threadIdx.x;
  const int lane = tid & 31, wid = tid >> 5;          // 32 waves
  const int half = wid & 1, nw = wid >> 1;            // 16 N-wave groups x 2 K-halves
  if (tid < KPAD) hbf[tid] = (__bf16)0.0f;
  if (tid < HID)  cst[tid] = 0.0f;
  __syncthreads();
  const int off   = (lane >= 16) ? 8 : 0;
  const int ktBeg = half ? 7 : 0;
  const int ktCnt = half ? (KT_H - 7) : 7;            // 6 or 7 (wave-uniform)

  for (int t = 0; t < TLEN; ++t) {
    if (tid < 100)  // pull this step's xg row toward the WGP while WMMAs run
      __builtin_prefetch(xg + (size_t)t * GATES + tid * 16, 0, 2);
    // Build this wave's A tiles from LDS h (replicated rows).
    v16bf aT[7];
#pragma unroll
    for (int q = 0; q < 7; ++q) {
      if (q < ktCnt) {
        const int kt = ktBeg + q;
        v8bf lo = *(const v8bf*)(&hbf[kt * 32 + off]);
        v8bf hi = *(const v8bf*)(&hbf[kt * 32 + off + 16]);
#pragma unroll
        for (int i = 0; i < 8; ++i) { aT[q][i] = lo[i]; aT[q][8 + i] = hi[i]; }
      }
    }
    // Uniform per-wave loops -> EXEC stays all-ones for WMMA.
    for (int nt = nw; nt < NT_G; nt += 16) {
      const __bf16* Btile = WhhP + ((size_t)nt * KT_H + ktBeg) * 512 + lane * 16;
      v8f acc = {};
#pragma unroll
      for (int q = 0; q < 7; ++q) {
        if (q < ktCnt) {
          v16bf b = *(const v16bf*)(Btile + (size_t)q * 512);
          acc = __builtin_amdgcn_wmma_f32_16x16x32_bf16(false, aT[q], false, b,
                                                        (short)0, acc, false, false);
        }
      }
      if (lane < 16) gatesP[half][nt * 16 + lane] = acc[0];   // row M=0
    }
    __syncthreads();
    if (tid < HID) {
      const float* xgt = xg + (size_t)t * GATES;
      const int j = tid;
      float gi = gatesP[0][j]           + gatesP[1][j]           + xgt[j]           + bhh[j];
      float gf = gatesP[0][HID + j]     + gatesP[1][HID + j]     + xgt[HID + j]     + bhh[HID + j];
      float gg = gatesP[0][2 * HID + j] + gatesP[1][2 * HID + j] + xgt[2 * HID + j] + bhh[2 * HID + j];
      float go = gatesP[0][3 * HID + j] + gatesP[1][3 * HID + j] + xgt[3 * HID + j] + bhh[3 * HID + j];
      float c = sigf(gf) * cst[j] + sigf(gi) * tanhf(gg);
      float h = sigf(go) * tanhf(c);
      cst[j] = c;
      __bf16 hb = (__bf16)h;
      hbf[j] = hb;
      outBF[(size_t)t * KPAD + j] = hb;
      if (outF32) outF32[(size_t)t * HID + j] = h;
    } else if (tid < KPAD) {
      outBF[(size_t)t * KPAD + tid] = (__bf16)0.0f;    // K padding
    }
    __syncthreads();
  }
}

// ---------------------------------------------------------------------------
// Action softmax(3) + push sigmoid(D.h): one wave per timestep, lane-strided
// loads with butterfly shuffle reduction (wave32).
// ---------------------------------------------------------------------------
__global__ void act_kernel(const float* __restrict__ h1, const float* __restrict__ actW,
                           const float* __restrict__ actb, const float* __restrict__ Dp,
                           float* __restrict__ probs, float* __restrict__ push) {
  const int gwid = (blockIdx.x * blockDim.x + threadIdx.x) >> 5;
  const int lane = threadIdx.x & 31;
  if (gwid >= TLEN) return;
  const float* h = h1 + (size_t)gwid * HID;
  float a0 = 0.f, a1 = 0.f, a2 = 0.f, dp = 0.f;
  for (int j = lane; j < HID; j += 32) {
    float hv = h[j];
    a0 += actW[j] * hv;
    a1 += actW[HID + j] * hv;
    a2 += actW[2 * HID + j] * hv;
    dp += Dp[j] * hv;
  }
#pragma unroll
  for (int s = 16; s > 0; s >>= 1) {
    a0 += __shfl_xor(a0, s);
    a1 += __shfl_xor(a1, s);
    a2 += __shfl_xor(a2, s);
    dp += __shfl_xor(dp, s);
  }
  if (lane == 0) {
    a0 += actb[0]; a1 += actb[1]; a2 += actb[2];
    float m = fmaxf(a0, fmaxf(a1, a2));
    float e0 = __expf(a0 - m), e1 = __expf(a1 - m), e2 = __expf(a2 - m);
    float s = e0 + e1 + e2;
    probs[gwid * 3 + 0] = e0 / s;
    probs[gwid * 3 + 1] = e1 / s;
    probs[gwid * 3 + 2] = e2 / s;
    push[gwid] = sigf(dp);
  }
}

// ---------------------------------------------------------------------------
// Stack-memory context, closed form. From the reference algebra:
// s_new[j] = s_old[j] for j>=1 (never written, init 0), and top reads only
// s_old[1] == 0, so context[t] = [probs0*push, 0, ..., 0]. Fully parallel
// float4 HBM stream (no sequential scan needed).
// ---------------------------------------------------------------------------
__global__ void stack_ctx_kernel(const float* __restrict__ probs,
                                 const float* __restrict__ push,
                                 float* __restrict__ ctx) {
  const size_t idx4 = (size_t)blockIdx.x * blockDim.x + threadIdx.x;
  const size_t total4 = (size_t)TLEN * STACKN * HID / 4;     // 13.1M float4
  if (idx4 >= total4) return;
  const int perT4 = STACKN * HID / 4;                        // 3200
  const int t   = (int)(idx4 / perT4);
  const int rem = (int)(idx4 % perT4);
  v4f v = {};
  if (rem < HID / 4) {                                       // row 0 of the stack
    float top = probs[t * 3] * push[t];
    v = (v4f){top, top, top, top};
  }
  *(v4f*)(ctx + idx4 * 4) = v;
}

// ---------------------------------------------------------------------------
// Key LSTM: single timestep with h_prev = c_prev = 0, then fc to 400 outputs.
// ---------------------------------------------------------------------------
__global__ void __launch_bounds__(1024)
key_kernel(const float* __restrict__ rh, const float* __restrict__ Wih,
           const float* __restrict__ bih, const float* __restrict__ bhh,
           const float* __restrict__ fcW, const float* __restrict__ fcb,
           float* __restrict__ out) {
  __shared__ float h[KEY];
  const int tid = threadIdx.x;
  if (tid < KEY) {
    float gi = bih[tid] + bhh[tid];
    float gg = bih[2 * KEY + tid] + bhh[2 * KEY + tid];
    float go = bih[3 * KEY + tid] + bhh[3 * KEY + tid];
    for (int j = 0; j < HID; ++j) {
      float x = rh[j];
      gi += Wih[(size_t)tid * HID + j] * x;
      gg += Wih[(size_t)(2 * KEY + tid) * HID + j] * x;
      go += Wih[(size_t)(3 * KEY + tid) * HID + j] * x;
    }
    float c = sigf(gi) * tanhf(gg);                 // f-gate * c_prev == 0
    h[tid] = sigf(go) * tanhf(c);
  }
  __syncthreads();
  if (tid < HID) {
    float acc = fcb[tid];
    for (int q = 0; q < KEY; ++q) acc += fcW[(size_t)tid * KEY + q] * h[q];
    out[tid] = acc;
  }
}

// ---------------------------------------------------------------------------
// Host launch
// ---------------------------------------------------------------------------
extern "C" void kernel_launch(void* const* d_in, const int* in_sizes, int n_in,
                              void* d_out, int out_size, void* d_ws, size_t ws_size,
                              hipStream_t stream) {
  const int*   xr       = (const int*)d_in[0];
  const int*   xcpc     = (const int*)d_in[1];
  const int*   xma      = (const int*)d_in[2];
  const float* emb_midi = (const float*)d_in[3];
  const float* emb_cpc  = (const float*)d_in[4];
  const float* emb_rhy  = (const float*)d_in[5];
  const float* Wih0     = (const float*)d_in[6];
  const float* Whh0     = (const float*)d_in[7];
  const float* bih0     = (const float*)d_in[8];
  const float* bhh0     = (const float*)d_in[9];
  const float* Wih1     = (const float*)d_in[10];
  const float* Whh1     = (const float*)d_in[11];
  const float* bih1     = (const float*)d_in[12];
  const float* bhh1     = (const float*)d_in[13];
  const float* fcW      = (const float*)d_in[14];
  const float* fcb      = (const float*)d_in[15];
  const float* actW     = (const float*)d_in[16];
  const float* actb     = (const float*)d_in[17];
  const float* Dp       = (const float*)d_in[18];
  const float* kWih     = (const float*)d_in[19];
  /* key_Whh (d_in[20]) multiplies h_prev = 0 */
  const float* kbih     = (const float*)d_in[21];
  const float* kbhh     = (const float*)d_in[22];
  const float* kfcW     = (const float*)d_in[23];
  const float* kfcb     = (const float*)d_in[24];

  char* ws = (char*)d_ws;
  size_t off = 0;
  auto alloc = [&](size_t bytes) { char* p = ws + off; off += (bytes + 255) & ~size_t(255); return p; };
  __bf16* xbf    = (__bf16*)alloc((size_t)TLEN * EMB3P * 2);
  __bf16* wih0p  = (__bf16*)alloc((size_t)NT_G * 5    * 512 * 2);
  __bf16* whh0p  = (__bf16*)alloc((size_t)NT_G * KT_H * 512 * 2);
  __bf16* wih1p  = (__bf16*)alloc((size_t)NT_G * KT_H * 512 * 2);
  __bf16* whh1p  = (__bf16*)alloc((size_t)NT_G * KT_H * 512 * 2);
  __bf16* fcwp   = (__bf16*)alloc((size_t)10   * KT_H * 512 * 2);
  float*  xg     = (float*)alloc((size_t)TLEN * GATES * 4);   // reused for both layers
  __bf16* h0bf   = (__bf16*)alloc((size_t)TLEN * KPAD * 2);
  __bf16* h1bf   = (__bf16*)alloc((size_t)TLEN * KPAD * 2);
  float*  h1f    = (float*)alloc((size_t)TLEN * HID * 4);
  float*  probs  = (float*)alloc((size_t)TLEN * 3 * 4);
  float*  push   = (float*)alloc((size_t)TLEN * 4);
  (void)ws_size; (void)n_in; (void)in_sizes; (void)out_size;

  float* out      = (float*)d_out;
  float* rtoken   = out;                       // [4096][150]
  float* ktoken   = out + (size_t)TLEN * EMB3; // [400]
  float* ctx      = ktoken + HID;              // [4096][32][400]

  // Pack weights into WMMA B layout (L2-resident thereafter).
  pack_b_kernel<<<dim3(NT_G, 5),    512, 0, stream>>>(Wih0, GATES, EMB3, 5,    wih0p);
  pack_b_kernel<<<dim3(NT_G, KT_H), 512, 0, stream>>>(Whh0, GATES, HID,  KT_H, whh0p);
  pack_b_kernel<<<dim3(NT_G, KT_H), 512, 0, stream>>>(Wih1, GATES, HID,  KT_H, wih1p);
  pack_b_kernel<<<dim3(NT_G, KT_H), 512, 0, stream>>>(Whh1, GATES, HID,  KT_H, whh1p);
  pack_b_kernel<<<dim3(10,   KT_H), 512, 0, stream>>>(fcW,  EMB3,  HID,  KT_H, fcwp);

  // Embedding gather + bf16 cast (padded K).
  embed_kernel<<<(TLEN * EMB3P + 255) / 256, 256, 0, stream>>>(
      xma, xcpc, xr, emb_midi, emb_cpc, emb_rhy, xbf);

  // xg0 = x @ Wih0^T + bih0   (WMMA GEMM, M=4096 N=1600 K=160)
  wmma_gemm_kernel<<<dim3(NT_G / 2, TLEN / 32), 32, 0, stream>>>(
      xbf, EMB3P, wih0p, bih0, xg, GATES, GATES, 5);

  // Layer-0 recurrence (persistent, single WGP).
  lstm_scan_kernel<<<1, 1024, 0, stream>>>(xg, whh0p, bhh0, h0bf, nullptr);

  // xg1 = h0 @ Wih1^T + bih1  (M=4096 N=1600 K=416)
  wmma_gemm_kernel<<<dim3(NT_G / 2, TLEN / 32), 32, 0, stream>>>(
      h0bf, KPAD, wih1p, bih1, xg, GATES, GATES, KT_H);

  // Layer-1 recurrence.
  lstm_scan_kernel<<<1, 1024, 0, stream>>>(xg, whh1p, bhh1, h1bf, h1f);

  // rtoken = h1 @ fcW^T + fcb (N=150, masked from 160)
  wmma_gemm_kernel<<<dim3(10 / 2, TLEN / 32), 32, 0, stream>>>(
      h1bf, KPAD, fcwp, fcb, rtoken, EMB3, EMB3, KT_H);

  // Action probabilities + push values (one wave per timestep).
  act_kernel<<<(TLEN * 32 + 255) / 256, 256, 0, stream>>>(h1f, actW, actb, Dp, probs, push);

  // Context output: closed-form parallel HBM stream.
  {
    size_t total4 = (size_t)TLEN * STACKN * HID / 4;
    stack_ctx_kernel<<<(unsigned)((total4 + 255) / 256), 256, 0, stream>>>(probs, push, ctx);
  }

  // Key LSTM single step + fc -> ktoken.
  key_kernel<<<1, 1024, 0, stream>>>(h1f + (size_t)(TLEN - 1) * HID,
                                     kWih, kbih, kbhh, kfcW, kfcb, ktoken);
}
